// CustomMaskedMHA_85590108274887
// MI455X (gfx1250) — matrix operand order repaired and checked by reading, not compiled, not run on hardware
//
#include <hip/hip_runtime.h>
#include <hip/hip_bf16.h>
#include <stdint.h>

// B=16, N=M=256, E=128, H=8, D=16, SCALE=4
typedef __bf16 bf16_t;
typedef __attribute__((ext_vector_type(16))) __bf16 v16bf;
typedef __attribute__((ext_vector_type(8)))  __bf16 v8bf;
typedef __attribute__((ext_vector_type(8)))  float  v8f;
typedef __attribute__((ext_vector_type(4)))  float  v4f;

#define INV_SCALE 0.25f

static __device__ __forceinline__ v16bf concat8(v8bf lo, v8bf hi) {
  return __builtin_shufflevector(lo, hi, 0,1,2,3,4,5,6,7,8,9,10,11,12,13,14,15);
}

// A fragment (16x32 bf16) from row-major [16][stride].
// Contiguous k order (B loaders match): lanes 0-15 row M=lane, K k0..k0+15; lanes 16-31 K k0+16..k0+31.
static __device__ __forceinline__ v16bf load_a_frag(const bf16_t* base, int stride, int k0, int lane) {
  int row = lane & 15;
  int kh  = (lane >> 4) << 4;
  const bf16_t* p = base + row * stride + k0 + kh;
  v8bf lo = *(const v8bf*)(p);
  v8bf hi = *(const v8bf*)(p + 8);
  return concat8(lo, hi);
}

// B fragment (32x16 bf16) from source stored as [col][k] (k contiguous).
// Contiguous k order (same as A): lane = column (mod 16); lanes 0-15 hold K k0..k0+15, lanes 16-31 K k0+16..k0+31.
static __device__ __forceinline__ v16bf load_b_frag_tr(const bf16_t* base, int stride, int k0, int lane) {
  int col = lane & 15;
  int ks  = k0 + ((lane >> 4) << 4);
  const bf16_t* p = base + col * stride + ks;
  v8bf lo = *(const v8bf*)(p);
  v8bf hi = *(const v8bf*)(p + 8);
  return concat8(lo, hi);
}

// B fragment gathered from an f32 row-major [k][ldc] weight matrix (L2-resident).
static __device__ __forceinline__ v16bf load_b_frag_w(const float* W, int ldc, int k0, int colBase, int lane) {
  int col = colBase + (lane & 15);
  int ks  = k0 + ((lane >> 4) << 4);
  v16bf b;
  #pragma unroll
  for (int j = 0; j < 16; ++j) b[j] = (bf16_t)W[(ks + j) * ldc + col];
  return b;
}

// Async global->LDS staging: one 16B chunk per calling thread (ASYNCcnt-tracked).
// lds_addr: low 32 bits of a generic pointer to __shared__ (== LDS byte offset per aperture rule).
static __device__ __forceinline__ void async_copy_b128(void* lds_dst, const void* gsrc, uint32_t goff) {
  uint32_t lds = (uint32_t)(uintptr_t)lds_dst;
  asm volatile("global_load_async_to_lds_b128 %0, %1, %2 offset:0"
               :: "v"(lds), "v"(goff), "s"(gsrc) : "memory");
}
static __device__ __forceinline__ void async_wait0() {
  asm volatile("s_wait_asynccnt 0x0" ::: "memory");
}

// ---------------- Kernel 1: X(rows x128) @ W(128x128) + bias, optional f32 / bf16 / bf16-transposed outs
__global__ __launch_bounds__(256) void proj_gemm(
    const float* __restrict__ X, const float* __restrict__ W, const float* __restrict__ bias,
    float* __restrict__ outF, bf16_t* __restrict__ outB, bf16_t* __restrict__ outBT) {
  __shared__ __align__(16) bf16_t sX[16 * 128];
  int tid = threadIdx.x, lane = tid & 31, w = tid >> 5;
  int row0 = blockIdx.x * 16;
  for (int i = tid; i < 16 * 128; i += 256)
    sX[i] = (bf16_t)X[(size_t)row0 * 128 + i];
  __syncthreads();
  int colBase = w * 16;
  v8f c = {};
  #pragma unroll
  for (int kk = 0; kk < 4; ++kk) {
    v16bf a = load_a_frag(sX, 128, kk * 32, lane);
    v16bf b = load_b_frag_w(W, 128, kk * 32, colBase, lane);
    c = __builtin_amdgcn_wmma_f32_16x16x32_bf16(false, a, false, b, (short)0, c, false, false);
  }
  int col = colBase + (lane & 15);
  int rsel = (lane >> 4) * 8;
  float bv = bias[col];
  #pragma unroll
  for (int r = 0; r < 8; ++r) {
    int row = row0 + r + rsel;
    float vo = c[r] + bv;
    if (outF) outF[(size_t)row * 128 + col] = vo;
    if (outB) outB[(size_t)row * 128 + col] = (bf16_t)vo;
    if (outBT) {                       // per-batch transposed: [b][col][m]
      int bb = row >> 8, mm = row & 255;
      outBT[((size_t)bb * 128 + col) * 256 + mm] = (bf16_t)vo;
    }
  }
}

// ---------------- Kernel 1b: build Acat = [A_q | T] (16x256 bf16 per (b,n), rows 8..15 zero), cq = q_h . bpe_h
__global__ __launch_bounds__(128) void build_acat(
    const float* __restrict__ qprojF, const float* __restrict__ Wpe, const float* __restrict__ bpe,
    bf16_t* __restrict__ acat, float* __restrict__ cq) {
  __shared__ float sW[128 * 128];
  __shared__ float sq[128];
  int tid = threadIdx.x;
  int bn = blockIdx.x;
  for (int i = tid; i < 128 * 128; i += 128) sW[i] = Wpe[i];
  sq[tid] = qprojF[(size_t)bn * 128 + tid];
  __syncthreads();
  bf16_t* arow = acat + (size_t)bn * (16 * 256);
  int e = tid;
  #pragma unroll
  for (int h = 0; h < 8; ++h) {
    float t = 0.f;
    #pragma unroll
    for (int d = 0; d < 16; ++d) t += sq[h * 16 + d] * sW[e * 128 + h * 16 + d];
    arow[h * 256 + 128 + e] = (bf16_t)t;                         // T[h][e]
    arow[h * 256 + e] = ((e >> 4) == h) ? (bf16_t)sq[e] : (bf16_t)0.f;  // block-diag q
  }
  for (int h = 8; h < 16; ++h) {
    arow[h * 256 + e] = (bf16_t)0.f;
    arow[h * 256 + 128 + e] = (bf16_t)0.f;
  }
  if (tid < 8) {
    float t = 0.f;
    for (int d = 0; d < 16; ++d) t += sq[tid * 16 + d] * bpe[tid * 16 + d];
    cq[(size_t)bn * 8 + tid] = t;
  }
}

// ---------------- Kernel 2: fused per-(b,n) attention with relative PE
__global__ __launch_bounds__(512) void attn_pe_main(
    const float* __restrict__ rel_pe, const float* __restrict__ attn_mask,
    const bf16_t* __restrict__ kprojB, const bf16_t* __restrict__ vprojT,
    const bf16_t* __restrict__ acat, const float* __restrict__ cq,
    const float* __restrict__ Wpe, const float* __restrict__ bpe,
    bf16_t* __restrict__ Zout) {
  __shared__ __align__(16) bf16_t sPT[128 * 256];   // 64KB rel_pe^T bf16 [e][m]
  __shared__ __align__(16) bf16_t sA[16 * 256];     // 8KB Acat; reused as R (16x128 bf16)
  __shared__ __align__(16) float  sS[16 * 256];     // 16KB scores; reused as U
  __shared__ __align__(16) bf16_t sAttn[16 * 256];  // 8KB attn bf16; reused as C3 (f32 16x128)
  int tid = threadIdx.x, lane = tid & 31, w = tid >> 5;
  int bn = blockIdx.x, bidx = bn >> 8;
  int l15 = lane & 15, sel = lane >> 4;
  int mt = w * 16;
  const float* relpe_bn = rel_pe + (size_t)bn * (256 * 128);
  { // stage Acat via async global->LDS DMA (8KB = 512 threads x 16B)
    const char* src = (const char*)(acat + (size_t)bn * (16 * 256));
    async_copy_b128((char*)sA + tid * 16, src, (uint32_t)tid * 16u);
    async_wait0();
  }
  __syncthreads();

  // ---- Phase A: S(16x256) = [A_q|T] @ [K^T ; P^T];  K=256 (128 kproj + 128 rel_pe)
  v8f c = {};
  const bf16_t* kp_b = kprojB + (size_t)bidx * (256 * 128) + mt * 128;
  #pragma unroll
  for (int kk = 0; kk < 8; ++kk) {
    v16bf a = load_a_frag(sA, 256, kk * 32, lane);
    v16bf b;
    if (kk < 4) {
      b = load_b_frag_tr(kp_b, 128, kk * 32, lane);   // qk part (L2-hot)
    } else {
      int mcol = mt + l15;
      int e0 = (kk - 4) * 32 + sel * 16;
      const float* pp = relpe_bn + (size_t)mcol * 128 + e0;  // 64B contiguous per lane
      __builtin_prefetch(pp + 512, 0, 1);
      v4f f0 = *(const v4f*)(pp);
      v4f f1 = *(const v4f*)(pp + 4);
      v4f f2 = *(const v4f*)(pp + 8);
      v4f f3 = *(const v4f*)(pp + 12);
      #pragma unroll
      for (int j = 0; j < 4; ++j) {
        b[j]      = (bf16_t)f0[j]; b[4 + j]  = (bf16_t)f1[j];
        b[8 + j]  = (bf16_t)f2[j]; b[12 + j] = (bf16_t)f3[j];
      }
      #pragma unroll
      for (int j = 0; j < 16; ++j) sPT[(e0 + j) * 256 + mcol] = b[j];  // park transposed for phase B
    }
    c = __builtin_amdgcn_wmma_f32_16x16x32_bf16(false, a, false, b, (short)0, c, false, false);
  }
  #pragma unroll
  for (int r = 0; r < 8; ++r) sS[(r + sel * 8) * 256 + mt + l15] = c[r];
  __syncthreads();

  // ---- Softmax over m (wave w = head h, wave32 shuffle reductions)
  if (w < 8) {
    int h = w;
    float cqv = cq[(size_t)bn * 8 + h];
    const float* mrow = attn_mask + (size_t)bn * 256;
    float vals[8]; float mx = -1e30f;
    #pragma unroll
    for (int i = 0; i < 8; ++i) {
      int ci = lane + 32 * i;
      float s = (sS[h * 256 + ci] + cqv) * INV_SCALE + mrow[ci];
      vals[i] = s; mx = fmaxf(mx, s);
    }
    #pragma unroll
    for (int off = 16; off > 0; off >>= 1) mx = fmaxf(mx, __shfl_xor(mx, off));
    float sum = 0.f;
    #pragma unroll
    for (int i = 0; i < 8; ++i) { vals[i] = __expf(vals[i] - mx); sum += vals[i]; }
    #pragma unroll
    for (int off = 16; off > 0; off >>= 1) sum += __shfl_xor(sum, off);
    float inv = 1.0f / sum;
    #pragma unroll
    for (int i = 0; i < 8; ++i) sAttn[h * 256 + lane + 32 * i] = (bf16_t)(vals[i] * inv);
  } else {
    #pragma unroll
    for (int i = 0; i < 8; ++i) sAttn[w * 256 + lane + 32 * i] = (bf16_t)0.f;
  }
  __syncthreads();

  // ---- Phase B: [R | outV'](16x256) = attn @ [P ; V];  K=256 over m
  v8f u = {};
  const bf16_t* vb = vprojT + ((size_t)bidx * 128 + (w - 8) * 16) * 256;
  #pragma unroll
  for (int kk = 0; kk < 8; ++kk) {
    v16bf a = load_a_frag(sAttn, 256, kk * 32, lane);
    v16bf b = (w < 8) ? load_b_frag_tr(sPT + mt * 256, 256, kk * 32, lane)
                      : load_b_frag_tr(vb, 256, kk * 32, lane);
    u = __builtin_amdgcn_wmma_f32_16x16x32_bf16(false, a, false, b, (short)0, u, false, false);
  }
  #pragma unroll
  for (int r = 0; r < 8; ++r) sS[(r + sel * 8) * 256 + mt + l15] = u[r];  // U
  if (w < 8) {
    bf16_t* R = sA;  // reuse: R[16][128] bf16
    #pragma unroll
    for (int r = 0; r < 8; ++r) R[(r + sel * 8) * 128 + mt + l15] = (bf16_t)u[r];
  }
  __syncthreads();

  // ---- GEMM3: C3 = R @ Wpe  (out_pe before diagonal-block selection)
  if (w < 8) {
    v8f c3 = {};
    #pragma unroll
    for (int kk = 0; kk < 4; ++kk) {
      v16bf a = load_a_frag(sA, 128, kk * 32, lane);
      v16bf b = load_b_frag_w(Wpe, 128, kk * 32, mt, lane);
      c3 = __builtin_amdgcn_wmma_f32_16x16x32_bf16(false, a, false, b, (short)0, c3, false, false);
    }
    float* C3 = (float*)sAttn;  // reuse: 16x128 f32
    #pragma unroll
    for (int r = 0; r < 8; ++r) C3[(r + sel * 8) * 128 + mt + l15] = c3[r];
  }
  __syncthreads();

  // ---- Combine: Z[e=(h,d)] = outV[h,d] + C3[h, h*16+d] + bpe (sum attn == 1)
  if (tid < 128) {
    int e = tid, h = e >> 4;
    float z = sS[h * 256 + 128 + e] + ((float*)sAttn)[h * 128 + e] + bpe[e];
    Zout[(size_t)bn * 128 + e] = (bf16_t)z;
  }
}

// ---------------- Kernel 3: OUT = Z @ Wo + bo
__global__ __launch_bounds__(256) void out_proj(
    const bf16_t* __restrict__ Z, const float* __restrict__ Wo, const float* __restrict__ bo,
    float* __restrict__ out) {
  __shared__ __align__(16) bf16_t sZ[16 * 128];
  int tid = threadIdx.x, lane = tid & 31, w = tid >> 5;
  int row0 = blockIdx.x * 16;
  { // stage Z via async global->LDS DMA (4KB = 256 threads x 16B)
    const char* src = (const char*)(Z + (size_t)row0 * 128);
    async_copy_b128((char*)sZ + tid * 16, src, (uint32_t)tid * 16u);
    async_wait0();
  }
  __syncthreads();
  int colBase = w * 16;
  v8f c = {};
  #pragma unroll
  for (int kk = 0; kk < 4; ++kk) {
    v16bf a = load_a_frag(sZ, 128, kk * 32, lane);
    v16bf b = load_b_frag_w(Wo, 128, kk * 32, colBase, lane);
    c = __builtin_amdgcn_wmma_f32_16x16x32_bf16(false, a, false, b, (short)0, c, false, false);
  }
  int col = colBase + (lane & 15);
  float bv = bo[col];
  #pragma unroll
  for (int r = 0; r < 8; ++r) {
    int row = row0 + r + (lane >> 4) * 8;
    out[(size_t)row * 128 + col] = c[r] + bv;
  }
}

extern "C" void kernel_launch(void* const* d_in, const int* in_sizes, int n_in,
                              void* d_out, int out_size, void* d_ws, size_t ws_size,
                              hipStream_t stream) {
  const float* query  = (const float*)d_in[0];
  const float* key    = (const float*)d_in[1];
  const float* value  = (const float*)d_in[2];
  const float* rel_pe = (const float*)d_in[3];
  const float* amask  = (const float*)d_in[4];
  const float* Wq  = (const float*)d_in[5];  const float* bq  = (const float*)d_in[6];
  const float* Wk  = (const float*)d_in[7];  const float* bk  = (const float*)d_in[8];
  const float* Wv  = (const float*)d_in[9];  const float* bv  = (const float*)d_in[10];
  const float* Wpe = (const float*)d_in[11]; const float* bpe = (const float*)d_in[12];
  const float* Wo  = (const float*)d_in[13]; const float* bo  = (const float*)d_in[14];
  (void)in_sizes; (void)n_in; (void)out_size; (void)ws_size;

  char* ws = (char*)d_ws;
  size_t off = 0;
  auto alloc = [&](size_t bytes) { void* p = ws + off; off += (bytes + 255) & ~(size_t)255; return p; };
  float*  qprojF = (float*) alloc((size_t)4096 * 128 * 4);   // q projection f32
  bf16_t* kprojB = (bf16_t*)alloc((size_t)4096 * 128 * 2);   // k projection bf16 [b][m][e]
  bf16_t* vprojT = (bf16_t*)alloc((size_t)4096 * 128 * 2);   // v projection bf16 transposed [b][e'][m]
  bf16_t* acat   = (bf16_t*)alloc((size_t)4096 * 16 * 256 * 2); // [A_q | T] per (b,n)
  float*  cqw    = (float*) alloc((size_t)4096 * 8 * 4);
  bf16_t* Zb     = (bf16_t*)alloc((size_t)4096 * 128 * 2);

  proj_gemm<<<256, 256, 0, stream>>>(query, Wq, bq, qprojF, nullptr, nullptr);
  proj_gemm<<<256, 256, 0, stream>>>(key,   Wk, bk, nullptr, kprojB, nullptr);
  proj_gemm<<<256, 256, 0, stream>>>(value, Wv, bv, nullptr, nullptr, vprojT);
  build_acat<<<4096, 128, 0, stream>>>(qprojF, Wpe, bpe, acat, cqw);
  attn_pe_main<<<4096, 512, 0, stream>>>(rel_pe, amask, kprojB, vprojT, acat, cqw, Wpe, bpe, Zb);
  out_proj<<<256, 256, 0, stream>>>(Zb, Wo, bo, (float*)d_out);
}
